// molan_model_GCN_84791244358234
// MI455X (gfx1250) — compile-verified, matrix operand order checked by reading, not run on hardware
//
#include <hip/hip_runtime.h>
#include <hip/hip_bf16.h>
#include <stdint.h>

// ---------------- constants (match reference) ----------------
#define N_NODES   50000
#define N_EDGES   600000
#define N_GRAPHS  2000
#define NODE_DIM  64
#define HID       128
#define MLP1_DIM  64
#define MLP2_DIM  32
#define N_CONV    3
#define BN_EPS    1e-5f

typedef __attribute__((ext_vector_type(16))) _Float16 v16h;
typedef __attribute__((ext_vector_type(8)))  _Float16 v8h;
typedef __attribute__((ext_vector_type(8)))  float    v8f;

typedef int v4i __attribute__((vector_size(16)));
typedef __attribute__((address_space(1))) v4i* as1_v4i;
typedef __attribute__((address_space(3))) v4i* as3_v4i;

// ----- gfx1250 async global->LDS copy (ASYNCcnt path), builtin if present, else inline asm -----
__device__ __forceinline__ void async_g2l_b128(const void* g, void* l) {
#if __has_builtin(__builtin_amdgcn_global_load_async_to_lds_b128)
    __builtin_amdgcn_global_load_async_to_lds_b128((as1_v4i)g, (as3_v4i)l, 0, 0);
#else
    uint32_t loff = (uint32_t)(uintptr_t)(as3_v4i)l;
    uint64_t gaddr = (uint64_t)(uintptr_t)g;
    asm volatile("global_load_async_to_lds_b128 %0, %1, off"
                 :: "v"(loff), "v"(gaddr) : "memory");
#endif
}

__device__ __forceinline__ void wait_asynccnt0() {
#if __has_builtin(__builtin_amdgcn_s_wait_asynccnt)
    __builtin_amdgcn_s_wait_asynccnt(0);
#else
    asm volatile("s_wait_asynccnt 0x0" ::: "memory");
#endif
}

// ---------------- elementwise helpers ----------------
__global__ void k_f32_to_f16(const float* __restrict__ in, _Float16* __restrict__ out, int n) {
    int i = blockIdx.x * blockDim.x + threadIdx.x;
    if (i < n) out[i] = (_Float16)in[i];
}

// W[K][N] fp32 (row-major) -> Wt[N][K] f16 (row-major), so each WMMA B-column is contiguous
__global__ void k_transpose_to_f16(const float* __restrict__ W, _Float16* __restrict__ Wt, int K, int N) {
    int i = blockIdx.x * blockDim.x + threadIdx.x;
    if (i < K * N) {
        int k = i / N, n = i % N;
        Wt[(size_t)n * K + k] = (_Float16)W[i];
    }
}

__global__ void k_fill(float* __restrict__ p, float v, int n) {
    int i = blockIdx.x * blockDim.x + threadIdx.x;
    if (i < n) p[i] = v;
}

// deg[dst] += 1 for each real edge (deg pre-filled with 1.0 for the self loop)
__global__ void k_degree(const int* __restrict__ ei, float* __restrict__ deg, int E) {
    int e = blockIdx.x * blockDim.x + threadIdx.x;
    if (e < E) atomicAdd(&deg[ei[E + e]], 1.0f);
}

__global__ void k_rsqrt_inplace(float* __restrict__ d, int n) {
    int i = blockIdx.x * blockDim.x + threadIdx.x;
    if (i < n) d[i] = rsqrtf(d[i]);   // deg >= 1 always (self loops)
}

// ---------------- WMMA fragment compute core ----------------
__device__ __forceinline__ v8f wmma_chunk(const _Float16* aptr, const _Float16* bptr, v8f acc) {
    v8h a0 = *(const v8h*)(aptr);
    v8h a1 = *(const v8h*)(aptr + 16);
    v8h b0 = *(const v8h*)(bptr);
    v8h b1 = *(const v8h*)(bptr + 16);
    v16h av, bv;
#pragma unroll
    for (int j = 0; j < 8; ++j) {
        av[j] = a0[j]; av[8 + j] = a1[j];
        bv[j] = b0[j]; bv[8 + j] = b1[j];
    }
    return __builtin_amdgcn_wmma_f32_16x16x32_f16(
        /*neg_a=*/false, av, /*neg_b=*/false, bv,
        /*c_mod=*/(short)0, acc, /*reuse_a=*/false, /*reuse_b=*/false);
}

// ---------------- LDS-staged WMMA GEMM for N==128 (big node GEMMs) ----------------
// One block (8 waves) = one 16-row M tile x all 8 N tiles. The 16xK A tile is a
// contiguous 16*K*2-byte region (whole rows) -> staged into LDS once via the
// gfx1250 async global->LDS path, then all 8 waves read fragments with ds_load.
__global__ void __launch_bounds__(256)
k_gemm_n128_wmma_lds(const _Float16* __restrict__ A,
                     const _Float16* __restrict__ Bt,
                     const float* __restrict__ bias,
                     float* __restrict__ C,
                     _Float16* __restrict__ C16,
                     int K, int relu) {
    __shared__ _Float16 shA[16 * 128];           // up to 4KB (K<=128)
    const int tid  = threadIdx.x;
    const int mt   = blockIdx.x;

    // --- stage A tile: 16*K halves, contiguous in global ---
    const int xfers = (16 * K * 2) / 16;         // 16B transfers: 128 (K=64) or 256 (K=128)
    const _Float16* gsrc = A + (size_t)mt * 16 * K;
    if (tid < xfers)
        async_g2l_b128(gsrc + tid * 8, shA + tid * 8);
    wait_asynccnt0();
    __syncthreads();

    // --- compute: wave w handles N-tile nt = w (N = 128 -> 8 tiles) ---
    const int nt   = tid >> 5;
    const int lane = tid & 31;
    const int r    = lane & 15;                  // A row / B col within tile
    const int hi   = lane >> 4;                  // K sub-block select

    const _Float16* arow = shA + r * K + hi * 8;
    const _Float16* brow = Bt + (size_t)(nt * 16 + r) * K + hi * 8;

    v8f acc = {};
    for (int kc = 0; kc < K; kc += 32)
        acc = wmma_chunk(arow + kc, brow + kc, acc);

    const int n = nt * 16 + r;
    const float bb = bias ? bias[n] : 0.0f;
#pragma unroll
    for (int i = 0; i < 8; ++i) {
        int m = mt * 16 + i + 8 * hi;            // lanes 16-31 hold rows M=i+8
        float v = acc[i] + bb;
        if (relu) v = fmaxf(v, 0.0f);
        C[(size_t)m * HID + n] = v;
        if (C16) C16[(size_t)m * HID + n] = (_Float16)v;
    }
}

// ---------------- direct-global WMMA GEMM (small MLP GEMMs) ----------------
__global__ void k_gemm_f16_wmma(const _Float16* __restrict__ A,
                                const _Float16* __restrict__ Bt,
                                const float* __restrict__ bias,
                                float* __restrict__ C,
                                _Float16* __restrict__ C16,
                                int M, int N, int K, int relu) {
    int wave   = blockIdx.x * (blockDim.x >> 5) + (threadIdx.x >> 5);
    int ntN    = N >> 4;
    int total  = (M >> 4) * ntN;
    if (wave >= total) return;              // wave-uniform guard: EXEC stays all-ones
    int mt = wave / ntN, nt = wave % ntN;
    int lane = threadIdx.x & 31;
    int r  = lane & 15;
    int hi = lane >> 4;

    const _Float16* arow = A  + (size_t)(mt * 16 + r) * K + hi * 8;
    const _Float16* brow = Bt + (size_t)(nt * 16 + r) * K + hi * 8;

    v8f acc = {};
    for (int kc = 0; kc < K; kc += 32)
        acc = wmma_chunk(arow + kc, brow + kc, acc);

    int n = nt * 16 + r;
    float bb = bias ? bias[n] : 0.0f;
#pragma unroll
    for (int i = 0; i < 8; ++i) {
        int m = mt * 16 + i + 8 * hi;
        float v = acc[i] + bb;
        if (relu) v = fmaxf(v, 0.0f);
        C[(size_t)m * N + n] = v;
        if (C16) C16[(size_t)m * N + n] = (_Float16)v;
    }
}

// ---------------- graph message passing ----------------
// out[dst] += h[src] * dis[src]*dis[dst]  (fp32 atomics; 25.6MB accumulator is L2-resident)
__global__ void k_scatter(const int* __restrict__ ei, const float* __restrict__ dis,
                          const float* __restrict__ h, float* __restrict__ out, int E) {
    int e = blockIdx.x;
    int f = threadIdx.x;
    int s = ei[e], d = ei[E + e];
    float nrm = dis[s] * dis[d];
    atomicAdd(&out[(size_t)d * HID + f], h[(size_t)s * HID + f] * nrm);
}

// self-loop contribution + bias + relu (in-place on the scatter accumulator)
__global__ void k_selfloop_bias_relu(float* __restrict__ x, const float* __restrict__ h,
                                     const float* __restrict__ dis, const float* __restrict__ b) {
    int n = blockIdx.x, f = threadIdx.x;
    size_t idx = (size_t)n * HID + f;
    float di = dis[n];
    float v = x[idx] + h[idx] * di * di + b[f];
    x[idx] = fmaxf(v, 0.0f);
}

// ---------------- BatchNorm (training-mode batch stats) ----------------
__global__ void k_bn_stats(const float* __restrict__ x, float* __restrict__ mu,
                           float* __restrict__ rstd, int Nn) {
    int f = blockIdx.x;                  // one block per feature (128)
    float s = 0.f, sq = 0.f;
    for (int n = threadIdx.x; n < Nn; n += blockDim.x) {
        float v = x[(size_t)n * HID + f];
        s += v; sq += v * v;
    }
    __shared__ float ss[256], s2[256];
    ss[threadIdx.x] = s; s2[threadIdx.x] = sq;
    __syncthreads();
    for (int o = blockDim.x >> 1; o > 0; o >>= 1) {
        if (threadIdx.x < o) { ss[threadIdx.x] += ss[threadIdx.x + o]; s2[threadIdx.x] += s2[threadIdx.x + o]; }
        __syncthreads();
    }
    if (threadIdx.x == 0) {
        float m   = ss[0] / (float)Nn;
        float var = s2[0] / (float)Nn - m * m;
        mu[f]   = m;
        rstd[f] = rsqrtf(var + BN_EPS);
    }
}

__global__ void k_bn_apply(float* __restrict__ x, _Float16* __restrict__ x16,
                           const float* __restrict__ mu, const float* __restrict__ rstd,
                           const float* __restrict__ gamma, const float* __restrict__ beta) {
    int n = blockIdx.x, f = threadIdx.x;
    size_t idx = (size_t)n * HID + f;
    float v = (x[idx] - mu[f]) * rstd[f] * gamma[f] + beta[f];
    x[idx]  = v;
    x16[idx] = (_Float16)v;
}

// ---------------- pooling + head ----------------
__global__ void k_pool(const float* __restrict__ emb, const int* __restrict__ batch,
                       float* __restrict__ g) {
    int n = blockIdx.x, f = threadIdx.x;
    atomicAdd(&g[(size_t)batch[n] * HID + f], emb[(size_t)n * HID + f]);
}

__global__ void k_pred(const float* __restrict__ h2, const float* __restrict__ w,
                       const float* __restrict__ b, float* __restrict__ out, int G) {
    int i = blockIdx.x * blockDim.x + threadIdx.x;
    if (i >= G * 2) return;
    int gi = i >> 1, c = i & 1;
    float s = b[c];
#pragma unroll
    for (int k = 0; k < MLP2_DIM; ++k) s += h2[(size_t)gi * MLP2_DIM + k] * w[k * 2 + c];
    out[i] = s;
}

// ---------------- host orchestration ----------------
static inline char* bump(char*& p, size_t bytes) {
    char* q = (char*)(((uintptr_t)p + 255) & ~(uintptr_t)255);
    p = q + bytes;
    return q;
}

extern "C" void kernel_launch(void* const* d_in, const int* in_sizes, int n_in,
                              void* d_out, int out_size, void* d_ws, size_t ws_size,
                              hipStream_t stream) {
    const float* x       = (const float*)d_in[0];
    const int*   ei      = (const int*)  d_in[1];
    const int*   batch   = (const int*)  d_in[2];
    const float* lin_w   = (const float*)d_in[3];
    const float* lin_b   = (const float*)d_in[4];
    const float* conv_w  = (const float*)d_in[5];
    const float* conv_b  = (const float*)d_in[6];
    const float* bn_g    = (const float*)d_in[7];
    const float* bn_b    = (const float*)d_in[8];
    const float* emb_w   = (const float*)d_in[9];
    const float* emb_b   = (const float*)d_in[10];
    const float* mlp1_w  = (const float*)d_in[11];
    const float* mlp1_b  = (const float*)d_in[12];
    const float* mlp2_w  = (const float*)d_in[13];
    const float* mlp2_b  = (const float*)d_in[14];
    const float* pred_w  = (const float*)d_in[15];
    const float* pred_b  = (const float*)d_in[16];
    float* out = (float*)d_out;

    // ---- workspace carve ----
    char* p = (char*)d_ws;
    _Float16* x0_16  = (_Float16*)bump(p, (size_t)N_NODES * NODE_DIM * 2);
    _Float16* x16    = (_Float16*)bump(p, (size_t)N_NODES * HID * 2);
    float*    xf     = (float*)   bump(p, (size_t)N_NODES * HID * 4);   // scatter accum / current features
    float*    hf     = (float*)   bump(p, (size_t)N_NODES * HID * 4);   // GEMM output
    float*    dis    = (float*)   bump(p, (size_t)N_NODES * 4);
    float*    mu     = (float*)   bump(p, HID * 4);
    float*    rstd   = (float*)   bump(p, HID * 4);
    _Float16* lin_wt = (_Float16*)bump(p, (size_t)NODE_DIM * HID * 2);
    _Float16* conv_wt= (_Float16*)bump(p, (size_t)N_CONV * HID * HID * 2);
    _Float16* emb_wt = (_Float16*)bump(p, (size_t)HID * HID * 2);
    _Float16* m1_wt  = (_Float16*)bump(p, (size_t)HID * MLP1_DIM * 2);
    _Float16* m2_wt  = (_Float16*)bump(p, (size_t)MLP1_DIM * MLP2_DIM * 2);
    float*    g      = (float*)   bump(p, (size_t)N_GRAPHS * HID * 4);
    _Float16* g16    = (_Float16*)bump(p, (size_t)N_GRAPHS * HID * 2);
    float*    h1f    = (float*)   bump(p, (size_t)N_GRAPHS * MLP1_DIM * 4);
    _Float16* h1_16  = (_Float16*)bump(p, (size_t)N_GRAPHS * MLP1_DIM * 2);
    float*    h2f    = (float*)   bump(p, (size_t)N_GRAPHS * MLP2_DIM * 4);
    (void)ws_size; (void)n_in; (void)in_sizes; (void)out_size;

    const int T = 256;
    auto cdiv = [](int a, int b) { return (a + b - 1) / b; };

    // ---- weight prep (f16, transposed to [N][K]) ----
    k_f32_to_f16<<<cdiv(N_NODES * NODE_DIM, T), T, 0, stream>>>(x, x0_16, N_NODES * NODE_DIM);
    k_transpose_to_f16<<<cdiv(NODE_DIM * HID, T), T, 0, stream>>>(lin_w, lin_wt, NODE_DIM, HID);
    for (int i = 0; i < N_CONV; ++i)
        k_transpose_to_f16<<<cdiv(HID * HID, T), T, 0, stream>>>(
            conv_w + (size_t)i * HID * HID, conv_wt + (size_t)i * HID * HID, HID, HID);
    k_transpose_to_f16<<<cdiv(HID * HID, T), T, 0, stream>>>(emb_w, emb_wt, HID, HID);
    k_transpose_to_f16<<<cdiv(HID * MLP1_DIM, T), T, 0, stream>>>(mlp1_w, m1_wt, HID, MLP1_DIM);
    k_transpose_to_f16<<<cdiv(MLP1_DIM * MLP2_DIM, T), T, 0, stream>>>(mlp2_w, m2_wt, MLP1_DIM, MLP2_DIM);

    // ---- degree / normalization:  dis = rsqrt(1 + in-degree) ----
    k_fill<<<cdiv(N_NODES, T), T, 0, stream>>>(dis, 1.0f, N_NODES);
    k_degree<<<cdiv(N_EDGES, T), T, 0, stream>>>(ei, dis, N_EDGES);
    k_rsqrt_inplace<<<cdiv(N_NODES, T), T, 0, stream>>>(dis, N_NODES);

    // ---- lin: x = x @ lin_w + lin_b  (no activation) ----
    k_gemm_n128_wmma_lds<<<N_NODES / 16, 256, 0, stream>>>(
        x0_16, lin_wt, lin_b, xf, x16, NODE_DIM, /*relu=*/0);

    // ---- 3 GCN conv layers ----
    for (int i = 0; i < N_CONV; ++i) {
        k_gemm_n128_wmma_lds<<<N_NODES / 16, 256, 0, stream>>>(
            x16, conv_wt + (size_t)i * HID * HID, nullptr, hf, nullptr, HID, /*relu=*/0);
        k_fill<<<cdiv(N_NODES * HID, T), T, 0, stream>>>(xf, 0.0f, N_NODES * HID);
        k_scatter<<<N_EDGES, HID, 0, stream>>>(ei, dis, hf, xf, N_EDGES);
        k_selfloop_bias_relu<<<N_NODES, HID, 0, stream>>>(xf, hf, dis, conv_b + (size_t)i * HID);
        k_bn_stats<<<HID, 256, 0, stream>>>(xf, mu, rstd, N_NODES);
        k_bn_apply<<<N_NODES, HID, 0, stream>>>(xf, x16, mu, rstd,
                                                bn_g + (size_t)i * HID, bn_b + (size_t)i * HID);
    }

    // ---- emb: relu(x @ emb_w + emb_b), then global_add_pool ----
    k_gemm_n128_wmma_lds<<<N_NODES / 16, 256, 0, stream>>>(
        x16, emb_wt, emb_b, hf, nullptr, HID, /*relu=*/1);
    k_fill<<<cdiv(N_GRAPHS * HID, T), T, 0, stream>>>(g, 0.0f, N_GRAPHS * HID);
    k_pool<<<N_NODES, HID, 0, stream>>>(hf, batch, g);
    k_f32_to_f16<<<cdiv(N_GRAPHS * HID, T), T, 0, stream>>>(g, g16, N_GRAPHS * HID);

    // ---- MLP head (direct-global WMMA) ----
    {
        int tiles = (N_GRAPHS / 16) * (MLP1_DIM / 16);
        k_gemm_f16_wmma<<<cdiv(tiles, 8), 256, 0, stream>>>(
            g16, m1_wt, mlp1_b, h1f, h1_16, N_GRAPHS, MLP1_DIM, HID, /*relu=*/1);
    }
    {
        int tiles = (N_GRAPHS / 16) * (MLP2_DIM / 16);
        k_gemm_f16_wmma<<<cdiv(tiles, 8), 256, 0, stream>>>(
            h1_16, m2_wt, mlp2_b, h2f, nullptr, N_GRAPHS, MLP2_DIM, MLP1_DIM, /*relu=*/1);
    }
    k_pred<<<cdiv(N_GRAPHS * 2, T), T, 0, stream>>>(h2f, pred_w, pred_b, out, N_GRAPHS);
}
